// MLAAttention_26594437497705
// MI455X (gfx1250) — matrix operand order repaired and checked by reading, not compiled, not run on hardware
//
#include <hip/hip_runtime.h>

// ---------------------------------------------------------------------------
// MLA attention prefill for gfx1250 (MI455X), wave32 + WMMA f16 (f32 accum).
// Shapes (fixed): B=2, S=2048, D_MODEL=2048, H=16, HD=128, LATENT=512.
// All operands converted once to f16 (working set ~106MB, fits 192MB L2);
// GEMMs stage LDS tiles with GLOBAL_LOAD_ASYNC_TO_LDS_B128 + double buffer.
// ---------------------------------------------------------------------------

typedef __attribute__((ext_vector_type(16))) _Float16 v16h;
typedef __attribute__((ext_vector_type(8)))  float    v8f;

constexpr int D_MODEL   = 2048;
constexpr int NUM_HEADS = 16;
constexpr int HEAD_DIM  = 128;
constexpr int LATENT    = 512;
constexpr int INNER     = NUM_HEADS * HEAD_DIM;   // 2048
constexpr int BATCH     = 2;
constexpr int SEQ       = 2048;
constexpr int MROWS     = BATCH * SEQ;            // 4096

#define WMMA_F16(a, b, c) \
  __builtin_amdgcn_wmma_f32_16x16x32_f16(false, (a), false, (b), (short)0, (c), false, false)

// A-fragment (lane = row M, K elems at kbase+{0..7} and kbase+16+{0..7})
__device__ __forceinline__ v16h frag_load_a(const _Float16* rowPtr, int kbase) {
  union { v16h h; uint4 u[2]; } f;
  f.u[0] = *(const uint4*)(rowPtr + kbase);
  f.u[1] = *(const uint4*)(rowPtr + kbase + 16);
  return f.h;
}

// B-fragment (lane = col N, K elems at kbase+{0..7} and kbase+16+{0..7}, row stride ld)
__device__ __forceinline__ v16h frag_load_b(const _Float16* p, int ld, int kbase) {
  v16h f;
#pragma unroll
  for (int s = 0; s < 8; ++s) {
    f[s]     = p[(size_t)(kbase + s) * ld];
    f[8 + s] = p[(size_t)(kbase + 16 + s) * ld];
  }
  return f;
}

// Per-lane async copy of 16 bytes global -> LDS (ASYNCcnt-tracked, GV mode).
__device__ __forceinline__ void async_copy_b128(const _Float16* gsrc, _Float16* ldst) {
  unsigned int  l = (unsigned int)(unsigned long long)ldst;  // flat LDS: offset in [31:0]
  unsigned long long g = (unsigned long long)gsrc;
  asm volatile("global_load_async_to_lds_b128 %0, %1, off" :: "v"(l), "v"(g) : "memory");
}

// ---------------------------------------------------------------------------
// f32 -> f16 conversion (vectorized, one-time pass per tensor)
// ---------------------------------------------------------------------------
__global__ __launch_bounds__(256) void cvt_f32_f16(const float* __restrict__ in,
                                                   _Float16* __restrict__ out, int n4) {
  int i = blockIdx.x * 256 + threadIdx.x;
  if (i < n4) {
    float4 v = ((const float4*)in)[i];
    union { _Float16 h[4]; uint2 u; } pk;
    pk.h[0] = (_Float16)v.x; pk.h[1] = (_Float16)v.y;
    pk.h[2] = (_Float16)v.z; pk.h[3] = (_Float16)v.w;
    ((uint2*)out)[i] = pk.u;
  }
}

// ---------------------------------------------------------------------------
// Tiled GEMM, all-f16 operands:  C[M,N] = A[M,K] * B[K,N]
// LDS double buffer, tiles staged with async-to-LDS b128 (4 ops/thread/tile),
// s_wait_asynccnt pipelining; 8 WMMA per wave per 32-deep K step.
// MODE 0: f32 row-major -> outF32                                  (y)
// MODE 1: f16 remapped [b,h,s,d]  (row=b*S+s, col=h*128+d)         (q, v)
// MODE 2: f16 remapped [b,h,d,s]  (per-head transpose)             (k^T)
// MODE 3: dual store f32 row-major + f16 row-major                 (c_kv)
// ---------------------------------------------------------------------------
template <int MODE>
__global__ __launch_bounds__(256) void mla_gemm(const _Float16* __restrict__ A,
                                                const _Float16* __restrict__ B,
                                                float* __restrict__ outF32,
                                                _Float16* __restrict__ outF16,
                                                int M, int N, int K) {
  constexpr int BM = 128, BN = 128, BK = 32;
  __shared__ __attribute__((aligned(16))) _Float16 Abuf[2][BM][BK];  // 2 x 8 KB
  __shared__ __attribute__((aligned(16))) _Float16 Bbuf[2][BK][BN];  // 2 x 8 KB

  const int tid   = threadIdx.x;
  const int lane  = tid & 31;
  const int wid   = tid >> 5;
  const int waveM = wid >> 2;          // 0..1  -> 64-row slab
  const int waveN = wid & 3;           // 0..3  -> 32-col slab
  const int l16   = lane & 15;
  const int lhi   = lane >> 4;
  const int kb    = lhi * 8;
  const int rowBase = blockIdx.y * BM;
  const int colBase = blockIdx.x * BN;

  // issue one K-tile's async copies: 2x A + 2x B 16-byte chunks per thread
  auto issue_tile = [&](int kk, int buf) {
#pragma unroll
    for (int p = 0; p < 2; ++p) {
      int idx = p * 256 + tid;                   // 0..511
      int r = idx >> 2, c8 = (idx & 3) * 8;      // A tile: 128 rows x 32 halfs
      async_copy_b128(A + (size_t)(rowBase + r) * K + kk + c8, &Abuf[buf][r][c8]);
    }
#pragma unroll
    for (int p = 0; p < 2; ++p) {
      int idx = p * 256 + tid;                   // 0..511
      int kr = idx >> 4, c8 = (idx & 15) * 8;    // B tile: 32 rows x 128 halfs
      async_copy_b128(B + (size_t)(kk + kr) * N + colBase + c8, &Bbuf[buf][kr][c8]);
    }
  };

  v8f acc[4][2];
#pragma unroll
  for (int mt = 0; mt < 4; ++mt)
#pragma unroll
    for (int nt = 0; nt < 2; ++nt)
#pragma unroll
      for (int e = 0; e < 8; ++e) acc[mt][nt][e] = 0.0f;

  const int nT = K / BK;
  issue_tile(0, 0);
  for (int t = 0; t < nT; ++t) {
    const int cur = t & 1;
    if (t + 1 < nT) {
      issue_tile((t + 1) * BK, cur ^ 1);
      // async ops complete in order: <=4 outstanding == tile t fully landed
      asm volatile("s_wait_asynccnt 0x4" ::: "memory");
    } else {
      asm volatile("s_wait_asynccnt 0x0" ::: "memory");
    }
    __syncthreads();

    v16h af[4], bf[2];
#pragma unroll
    for (int mt = 0; mt < 4; ++mt)
      af[mt] = frag_load_a(&Abuf[cur][waveM * 64 + mt * 16 + l16][0], kb);
#pragma unroll
    for (int nt = 0; nt < 2; ++nt)
      bf[nt] = frag_load_b(&Bbuf[cur][0][waveN * 32 + nt * 16 + l16], BN, kb);
#pragma unroll
    for (int mt = 0; mt < 4; ++mt)
#pragma unroll
      for (int nt = 0; nt < 2; ++nt)
        acc[mt][nt] = WMMA_F16(af[mt], bf[nt], acc[mt][nt]);
    __syncthreads();
  }

  // ---- epilogue: D layout = VGPR i -> row i+8*lhi, col l16 ----
#pragma unroll
  for (int mt = 0; mt < 4; ++mt)
#pragma unroll
    for (int nt = 0; nt < 2; ++nt)
#pragma unroll
      for (int i = 0; i < 8; ++i) {
        int r = rowBase + waveM * 64 + mt * 16 + i + lhi * 8;
        int c = colBase + waveN * 32 + nt * 16 + l16;
        float val = acc[mt][nt][i];
        if constexpr (MODE == 0) {
          outF32[(size_t)r * N + c] = val;
        } else if constexpr (MODE == 3) {
          outF32[(size_t)r * N + c] = val;
          outF16[(size_t)r * N + c] = (_Float16)val;
        } else {
          int b = r >> 11, s = r & (SEQ - 1);
          int h = c >> 7, d = c & (HEAD_DIM - 1);
          if constexpr (MODE == 1)
            outF16[(((size_t)(b * NUM_HEADS + h)) * SEQ + s) * HEAD_DIM + d] = (_Float16)val;
          else
            outF16[(((size_t)(b * NUM_HEADS + h)) * HEAD_DIM + d) * SEQ + s] = (_Float16)val;
        }
      }
}

// ---------------------------------------------------------------------------
// Flash attention (causal).  grid = (SEQ/128, H, B), block = 256 (8 waves).
// Each wave owns 16 query rows: Q A-frags resident in VGPRs, loop over 32-key
// chunks up to the causal limit; scores via 4 WMMA (head-dim), online softmax
// with shfl_xor(width=16) row reductions, P bounced through per-wave LDS to
// re-shape D-layout -> A-layout, PV via 8 WMMA per chunk.
//   q : f16 [B,H,S,HD]   kt : f16 [B,H,HD,S]   v : f16 [B,H,S,HD]
//   o : f16 [B,S,H*HD]  (row-major [4096 x 2048], ready as A of y = o @ w_o)
// ---------------------------------------------------------------------------
__global__ __launch_bounds__(256) void mla_attn(const _Float16* __restrict__ q,
                                                const _Float16* __restrict__ kt,
                                                const _Float16* __restrict__ v,
                                                _Float16* __restrict__ o) {
  __shared__ __attribute__((aligned(16))) _Float16 Ps[8][16][32];  // 8 KB

  const int tid  = threadIdx.x;
  const int lane = tid & 31;
  const int wid  = tid >> 5;
  const int l16  = lane & 15;
  const int lhi  = lane >> 4;
  const int kb   = lhi * 8;
  const int b    = blockIdx.z;
  const int h    = blockIdx.y;
  const int qbase = (blockIdx.x * 8 + wid) * 16;

  const _Float16* qp = q  + ((size_t)(b * NUM_HEADS + h)) * SEQ * HEAD_DIM;
  const _Float16* kp = kt + ((size_t)(b * NUM_HEADS + h)) * HEAD_DIM * SEQ;
  const _Float16* vp = v  + ((size_t)(b * NUM_HEADS + h)) * SEQ * HEAD_DIM;

  // Q fragments: 16 rows x 128 head-dim = 4 A-frags, resident
  v16h qf[4];
  {
    const _Float16* qrow = qp + (size_t)(qbase + l16) * HEAD_DIM;
#pragma unroll
    for (int ks = 0; ks < 4; ++ks) qf[ks] = frag_load_a(qrow, ks * 32 + kb);
  }

  v8f yacc[8];
#pragma unroll
  for (int nt = 0; nt < 8; ++nt)
#pragma unroll
    for (int e = 0; e < 8; ++e) yacc[nt][e] = 0.0f;

  float rmax[8], rsum[8];
#pragma unroll
  for (int i = 0; i < 8; ++i) { rmax[i] = -1e30f; rsum[i] = 0.0f; }

  const float scale = 0.08838834764831845f;  // 1/sqrt(128)
  const int nchunks = (qbase + 47) >> 5;     // causal: keys <= qbase+15

  for (int kc = 0; kc < nchunks; ++kc) {
    // ---- scores: S = Q * K^T over 32 keys (2 n-tiles x 4 head-dim steps) ----
    v8f sacc[2];
#pragma unroll
    for (int t = 0; t < 2; ++t)
#pragma unroll
      for (int e = 0; e < 8; ++e) sacc[t][e] = 0.0f;
#pragma unroll
    for (int t = 0; t < 2; ++t)
#pragma unroll
      for (int ks = 0; ks < 4; ++ks) {
        // B-frag: lane = key column l16, 16 head-dim rows of K^T (table order)
        v16h kf = frag_load_b(kp + (size_t)(ks * 32) * SEQ + kc * 32 + t * 16 + l16, SEQ, kb);
        sacc[t] = WMMA_F16(qf[ks], kf, sacc[t]);
      }

    // ---- online softmax + stage P into per-wave LDS ----
#pragma unroll
    for (int i = 0; i < 8; ++i) {
      int qrow = qbase + i + lhi * 8;
      int k0 = kc * 32 + l16, k1 = k0 + 16;
      float s0 = sacc[0][i] * scale;
      float s1 = sacc[1][i] * scale;
      if (k0 > qrow) s0 = -1e30f;
      if (k1 > qrow) s1 = -1e30f;
      float mloc = fmaxf(s0, s1);
      mloc = fmaxf(mloc, __shfl_xor(mloc, 1, 16));
      mloc = fmaxf(mloc, __shfl_xor(mloc, 2, 16));
      mloc = fmaxf(mloc, __shfl_xor(mloc, 4, 16));
      mloc = fmaxf(mloc, __shfl_xor(mloc, 8, 16));
      float mnew  = fmaxf(rmax[i], mloc);
      float alpha = __expf(rmax[i] - mnew);
      rmax[i] = mnew;
      float p0 = __expf(s0 - mnew);
      float p1 = __expf(s1 - mnew);
      float ps = p0 + p1;
      ps += __shfl_xor(ps, 1, 16);
      ps += __shfl_xor(ps, 2, 16);
      ps += __shfl_xor(ps, 4, 16);
      ps += __shfl_xor(ps, 8, 16);
      rsum[i] = rsum[i] * alpha + ps;
#pragma unroll
      for (int nt = 0; nt < 8; ++nt) yacc[nt][i] *= alpha;
      int prow = i + lhi * 8;                 // D-layout row -> LDS row
      Ps[wid][prow][l16]      = (_Float16)p0;
      Ps[wid][prow][16 + l16] = (_Float16)p1;
    }
    // Same-wave DS ops are in-order; fence the compiler + wait out the writes.
    asm volatile("s_wait_dscnt 0" ::: "memory");

    // ---- y += P * V (A-frag from LDS, B-frags straight from v[b,h,s,d]) ----
    v16h pf = frag_load_a(&Ps[wid][l16][0], kb);
#pragma unroll
    for (int nt = 0; nt < 8; ++nt) {
      v16h vf = frag_load_b(vp + (size_t)(kc * 32) * HEAD_DIM + nt * 16 + l16, HEAD_DIM, kb);
      yacc[nt] = WMMA_F16(pf, vf, yacc[nt]);
    }
  }

  // ---- normalize + store o as f16 row-major [B*S, H*HD] ----
#pragma unroll
  for (int i = 0; i < 8; ++i) {
    float inv = 1.0f / rsum[i];
    int s = qbase + i + lhi * 8;
    size_t rowoff = ((size_t)(b * SEQ + s)) * INNER + h * HEAD_DIM;
#pragma unroll
    for (int nt = 0; nt < 8; ++nt)
      o[rowoff + nt * 16 + l16] = (_Float16)(yacc[nt][i] * inv);
  }
}

// ---------------------------------------------------------------------------
// Launch (graph-capture safe: all on `stream`, no alloc/sync).
// ws layout (f16): x | w_q | w_kvd | w_kup | w_vup | w_o | c_kv | q | k^T | v | o
// d_out: y f32 [2,2048,2048] then c_kv f32 [2,2048,512].
// ---------------------------------------------------------------------------
extern "C" void kernel_launch(void* const* d_in, const int* in_sizes, int n_in,
                              void* d_out, int out_size, void* d_ws, size_t ws_size,
                              hipStream_t stream) {
  (void)in_sizes; (void)n_in; (void)out_size; (void)ws_size;
  const float* x     = (const float*)d_in[0];
  const float* w_q   = (const float*)d_in[1];
  const float* w_kvd = (const float*)d_in[2];
  const float* w_kup = (const float*)d_in[3];
  const float* w_vup = (const float*)d_in[4];
  const float* w_o   = (const float*)d_in[5];

  float* y_out   = (float*)d_out;
  float* ckv_out = y_out + (size_t)MROWS * D_MODEL;   // second tuple output

  _Float16* p = (_Float16*)d_ws;
  _Float16* x16    = p; p += (size_t)MROWS * D_MODEL;        // 8.4M
  _Float16* wq16   = p; p += (size_t)D_MODEL * INNER;        // 4.2M
  _Float16* wkvd16 = p; p += (size_t)D_MODEL * LATENT;       // 1.0M
  _Float16* wkup16 = p; p += (size_t)LATENT * INNER;         // 1.0M
  _Float16* wvup16 = p; p += (size_t)LATENT * INNER;         // 1.0M
  _Float16* wo16   = p; p += (size_t)INNER * D_MODEL;        // 4.2M
  _Float16* ckv16  = p; p += (size_t)MROWS * LATENT;         // 2.1M
  _Float16* qb     = p; p += (size_t)MROWS * INNER;          // 8.4M
  _Float16* kt     = p; p += (size_t)MROWS * INNER;          // 8.4M
  _Float16* vb     = p; p += (size_t)MROWS * INNER;          // 8.4M
  _Float16* ob     = p;                                      // 8.4M

  dim3 blk(256);
  auto cvt = [&](const float* in, _Float16* out, size_t n) {
    int n4 = (int)(n / 4);
    cvt_f32_f16<<<dim3((n4 + 255) / 256), blk, 0, stream>>>(in, out, n4);
  };
  // 0) one-time f32 -> f16 down-conversion of all GEMM operands
  cvt(x,     x16,    (size_t)MROWS * D_MODEL);
  cvt(w_q,   wq16,   (size_t)D_MODEL * INNER);
  cvt(w_kvd, wkvd16, (size_t)D_MODEL * LATENT);
  cvt(w_kup, wkup16, (size_t)LATENT * INNER);
  cvt(w_vup, wvup16, (size_t)LATENT * INNER);
  cvt(w_o,   wo16,   (size_t)INNER * D_MODEL);

  // 1) c_kv = x @ w_kv_down -> f32 (output #2) + f16 (A of k/v GEMMs)
  mla_gemm<3><<<dim3(LATENT / 128, MROWS / 128), blk, 0, stream>>>(
      x16, wkvd16, ckv_out, ckv16, MROWS, LATENT, D_MODEL);
  // 2) q = x @ w_q -> f16 [b,h,s,d]
  mla_gemm<1><<<dim3(INNER / 128, MROWS / 128), blk, 0, stream>>>(
      x16, wq16, (float*)nullptr, qb, MROWS, INNER, D_MODEL);
  // 3) k^T = (c_kv @ w_k_up)^T -> f16 [b,h,d,s]
  mla_gemm<2><<<dim3(INNER / 128, MROWS / 128), blk, 0, stream>>>(
      ckv16, wkup16, (float*)nullptr, kt, MROWS, INNER, LATENT);
  // 4) v = c_kv @ w_v_up -> f16 [b,h,s,d]
  mla_gemm<1><<<dim3(INNER / 128, MROWS / 128), blk, 0, stream>>>(
      ckv16, wvup16, (float*)nullptr, vb, MROWS, INNER, LATENT);
  // 5) causal flash attention -> o f16 [b,s,h*d]
  mla_attn<<<dim3(SEQ / 128, NUM_HEADS, BATCH), blk, 0, stream>>>(qb, kt, vb, ob);
  // 6) y = o @ w_o -> f32 d_out
  mla_gemm<0><<<dim3(D_MODEL / 128, MROWS / 128), blk, 0, stream>>>(
      ob, wo16, y_out, (_Float16*)nullptr, MROWS, D_MODEL, INNER);
}
